// FourierKANLayer_54674933678526
// MI455X (gfx1250) — matrix operand order repaired and checked by reading, not compiled, not run on hardware
//
#include <hip/hip_runtime.h>

// FourierKAN as a WMMA GEMM: out[4096,512] = T[4096,32768] x W[32768,512] + bias
// T computed on the fly (trig recurrence), W = fourier_coeffs flattened.
// Double-buffered LDS pipeline, packed f16 conversions, 16B-aligned LDS rows
// so fragment loads lower to ds_load_b128.

typedef __attribute__((ext_vector_type(16))) _Float16 v16h;
typedef __attribute__((ext_vector_type(2)))  __fp16   v2hf;   // cvt_pkrtz return type
typedef __attribute__((ext_vector_type(8)))  float    v8f;

#define IN_DIM   512
#define OUT_DIM  512
#define NHARM    32
#define BATCH    4096

#define TM 64            // output tile rows per workgroup
#define TN 64            // output tile cols per workgroup
#define KC 64            // K chunk = one input column (32 harmonics x {cos,sin})
#define LDU 36           // LDS row stride in uints: multiple of 4 -> b128-mergeable

union FragAB { unsigned int u[8]; v16h h; };
union P2     { v2hf h2; unsigned int u; };

struct Stage {           // register-staged global data for one K chunk
    float  xv;
    float4 w4[8];        // pairs: w4[2p] = row k0, w4[2p+1] = row k0+1
};

__global__ __launch_bounds__(128) void fourier_kan_wmma(
    const float* __restrict__ x,     // [BATCH, IN_DIM]
    const float* __restrict__ W,     // [IN_DIM*NHARM*2, OUT_DIM]
    const float* __restrict__ bias,  // [OUT_DIM]
    float* __restrict__ out)         // [BATCH, OUT_DIM]
{
    __shared__ unsigned int ldsA[2][TM * LDU]; // trig features [b][k], packed halves
    __shared__ unsigned int ldsB[2][TN * LDU]; // weights transposed [n][k], packed halves

    const int tid   = threadIdx.x;
    const int lane  = tid & 31;          // wave32
    const int wave  = tid >> 5;          // 0..3
    const int waveM = (wave & 1) * 32;
    const int waveN = (wave >> 1) * 32;

    const int tileN = blockIdx.x * TN;
    const int tileM = blockIdx.y * TM;

    const int  l15  = lane & 15;
    const bool hi   = lane >= 16;
    const int  off4 = hi ? 4 : 0;        // A-fragment uint offset (ISA layout)
    const int  off8 = hi ? 8 : 0;        // unused; B fragment also uses off4

    // trig fill mapping: 2 threads per batch row
    const int bLoc = tid >> 1;           // 0..63
    const int g0   = (tid & 1) * 16;     // harmonics [0..15] or [16..31]

    // weight fill mapping: each thread owns k-row pairs (k0,k0+1), float4 over N
    const int kHalf = (tid >> 4) * 2;    // 0,2,..,14
    const int n4    = (tid & 15) * 4;    // 0..60

    v8f acc[2][2];
#pragma unroll
    for (int mi = 0; mi < 2; ++mi)
#pragma unroll
        for (int ni = 0; ni < 2; ++ni)
            acc[mi][ni] = (v8f){0.f,0.f,0.f,0.f,0.f,0.f,0.f,0.f};

    auto stage_load = [&](Stage& st, int i) {
        st.xv = x[(size_t)(tileM + bLoc) * IN_DIM + i];
        const float* Wbase = W + (size_t)i * KC * OUT_DIM + tileN;
#pragma unroll
        for (int p = 0; p < 4; ++p) {
            const int k0 = p * 16 + kHalf;
            st.w4[2 * p]     = *(const float4*)(Wbase + (size_t)k0 * OUT_DIM + n4);
            st.w4[2 * p + 1] = *(const float4*)(Wbase + (size_t)(k0 + 1) * OUT_DIM + n4);
        }
    };

    auto stage_store = [&](const Stage& st, unsigned int* A, unsigned int* B) {
        // trig features: one sincos seed + angle-addition recurrence, pk-converted.
        // 16 consecutive aligned uints per thread -> merges into ds_store_b128s.
        float c1, s1;  __sincosf(st.xv, &s1, &c1);
        float c, s;    __sincosf((float)(g0 + 1) * st.xv, &s, &c);
        unsigned int* rowA = &A[bLoc * LDU + g0];
        unsigned int  trig[16];
#pragma unroll
        for (int g = 0; g < 16; ++g) {
            P2 p; p.h2 = __builtin_amdgcn_cvt_pkrtz(c, s);   // halves k=2g, 2g+1
            trig[g] = p.u;
            const float cn = c * c1 - s * s1;
            const float sn = s * c1 + c * s1;
            c = cn; s = sn;
        }
#pragma unroll
        for (int g = 0; g < 16; ++g) rowA[g] = trig[g];
        // weights: pk-convert adjacent k pair -> one aligned uint in [n][k] layout
#pragma unroll
        for (int p = 0; p < 4; ++p) {
            const int k0 = p * 16 + kHalf;
            const int ku = k0 >> 1;
            const float4 r0 = st.w4[2 * p];
            const float4 r1 = st.w4[2 * p + 1];
            P2 t;
            t.h2 = __builtin_amdgcn_cvt_pkrtz(r0.x, r1.x); B[(n4 + 0) * LDU + ku] = t.u;
            t.h2 = __builtin_amdgcn_cvt_pkrtz(r0.y, r1.y); B[(n4 + 1) * LDU + ku] = t.u;
            t.h2 = __builtin_amdgcn_cvt_pkrtz(r0.z, r1.z); B[(n4 + 2) * LDU + ku] = t.u;
            t.h2 = __builtin_amdgcn_cvt_pkrtz(r0.w, r1.w); B[(n4 + 3) * LDU + ku] = t.u;
        }
    };

    auto mma_step = [&](const unsigned int* A, const unsigned int* B) {
#pragma unroll
        for (int kk = 0; kk < 2; ++kk) {        // two K sub-steps of 32
            const int kb = kk * 16;
            FragAB afr[2], bfr[2];
#pragma unroll
            for (int mi = 0; mi < 2; ++mi) {
                const unsigned int* pa = &A[(waveM + mi * 16 + l15) * LDU + kb];
#pragma unroll
                for (int j = 0; j < 4; ++j) {   // ISA A layout: K interleave
                    afr[mi].u[j]     = pa[j + off4];
                    afr[mi].u[j + 4] = pa[j + 8 + off4];
                }
            }
#pragma unroll
            for (int ni = 0; ni < 2; ++ni) {
                const unsigned int* pb = &B[(waveN + ni * 16 + l15) * LDU + kb];
#pragma unroll
                for (int j = 0; j < 4; ++j) {   // B layout matches A: K interleave
                    bfr[ni].u[j]     = pb[j + off4];
                    bfr[ni].u[j + 4] = pb[j + 8 + off4];
                }
            }
#pragma unroll
            for (int mi = 0; mi < 2; ++mi)
#pragma unroll
                for (int ni = 0; ni < 2; ++ni)
                    acc[mi][ni] = __builtin_amdgcn_wmma_f32_16x16x32_f16(
                        false, afr[mi].h, false, bfr[ni].h,
                        (short)0, acc[mi][ni], false, false);
        }
    };

    // ---- pipeline prologue: fill buffer 0 ----
    Stage st;
    stage_load(st, 0);
    stage_store(st, ldsA[0], ldsB[0]);
    __syncthreads();

    // ---- main pipeline: one barrier per K chunk ----
    int buf = 0;
    for (int i = 0; i < IN_DIM; ++i) {
        const bool hasNext = (i + 1) < IN_DIM;
        if (hasNext) {
            stage_load(st, i + 1);               // issue globals early (overlap WMMA)
            if (tid == 0 && i + 2 < IN_DIM)      // warm L2 one chunk further out
                __builtin_prefetch(W + (size_t)(i + 2) * KC * OUT_DIM + tileN, 0, 1);
        }

        mma_step(ldsA[buf], ldsB[buf]);          // matrix math on current buffer

        if (hasNext)
            stage_store(st, ldsA[buf ^ 1], ldsB[buf ^ 1]);  // fill other buffer

        __syncthreads();
        buf ^= 1;
    }

    // ---- epilogue: D layout (VGPR r -> M=r, lanes 16-31 -> M=r+8), add bias ----
    const int rowBase = tileM + waveM + (hi ? 8 : 0);
#pragma unroll
    for (int ni = 0; ni < 2; ++ni) {
        const int col = tileN + waveN + ni * 16 + l15;
        const float bv = bias[col];
#pragma unroll
        for (int mi = 0; mi < 2; ++mi)
#pragma unroll
            for (int r = 0; r < 8; ++r)
                out[(size_t)(rowBase + mi * 16 + r) * OUT_DIM + col] = acc[mi][ni][r] + bv;
    }
}

extern "C" void kernel_launch(void* const* d_in, const int* in_sizes, int n_in,
                              void* d_out, int out_size, void* d_ws, size_t ws_size,
                              hipStream_t stream) {
    const float* x    = (const float*)d_in[0];
    const float* W    = (const float*)d_in[1];  // fourier_coeffs [512,32,2,512] flat
    const float* bias = (const float*)d_in[2];
    float* out        = (float*)d_out;

    dim3 grid(OUT_DIM / TN, BATCH / TM);   // 8 x 64 = 512 workgroups
    fourier_kan_wmma<<<grid, 128, 0, stream>>>(x, W, bias, out);
}
